// TAGM_Bi_15247133901425
// MI455X (gfx1250) — matrix operand order repaired and checked by reading, not compiled, not run on hardware
//
#include <hip/hip_runtime.h>
#include <hip/hip_bf16.h>
#include <cstdint>

// Problem dims (fixed by the reference)
static constexpr int kB = 64, kT = 512, kI = 512, kH = 1024, kC = 1024;
static constexpr float kSigW = 3.0f;

typedef __bf16 bf16_t;
typedef __attribute__((ext_vector_type(16))) __bf16 v16bf;
typedef __attribute__((ext_vector_type(8)))  float  v8f;

union FragU { v16bf v; unsigned int u[8]; };

__device__ __forceinline__ unsigned short f2bf(float f) {
  unsigned u = __float_as_uint(f);
  u += 0x7FFFu + ((u >> 16) & 1u);            // round-to-nearest-even
  return (unsigned short)(u >> 16);
}

// A-matrix fragment (16x32 bf16, row-major source): per ISA 7.12.2,
// lanes 0-15 hold K={0..7,16..23}, lanes 16-31 hold K={8..15,24..31}, 2 per VGPR.
__device__ __forceinline__ v16bf load_fragA(const bf16_t* rowptr, int k0, int half) {
  FragU f;
#pragma unroll
  for (int v = 0; v < 8; ++v) {
    int k = k0 + ((v < 4) ? (2 * v) : (16 + 2 * (v - 4))) + half * 8;
    f.u[v] = *reinterpret_cast<const unsigned int*>(rowptr + k);
  }
  return f.v;
}

// B-matrix fragment (32x16 bf16): lane holds column n; lanes 0-15 K={0..7,16..23},
// lanes 16-31 K={8..15,24..31} (2 consecutive K per VGPR). rowptr = row n of W (W[n,k]).
__device__ __forceinline__ v16bf load_fragB(const bf16_t* rowptr, int k0, int half) {
  FragU f;
#pragma unroll
  for (int v = 0; v < 8; ++v) {
    int k = k0 + ((v < 4) ? (2 * v) : (16 + 2 * (v - 4))) + half * 8;
    f.u[v] = *reinterpret_cast<const unsigned int*>(rowptr + k);
  }
  return f.v;
}

__device__ __forceinline__ v8f wmma_bf16(v16bf a, v16bf b, v8f c) {
  return __builtin_amdgcn_wmma_f32_16x16x32_bf16(false, a, false, b, (short)0, c, false, false);
}

__device__ __forceinline__ float sigmf(float x) { return 1.0f / (1.0f + __expf(-x)); }

// Sense-free generation barrier across nblk blocks (all resident: <=32 blocks).
__device__ __forceinline__ void grid_barrier(unsigned* cnt, unsigned* gen, unsigned nblk) {
  __syncthreads();
  if (threadIdx.x == 0) {
    __threadfence();
    unsigned g = __hip_atomic_load(gen, __ATOMIC_ACQUIRE, __HIP_MEMORY_SCOPE_AGENT);
    unsigned prev = __hip_atomic_fetch_add(cnt, 1u, __ATOMIC_ACQ_REL, __HIP_MEMORY_SCOPE_AGENT);
    if (prev == nblk - 1u) {
      __hip_atomic_store(cnt, 0u, __ATOMIC_RELAXED, __HIP_MEMORY_SCOPE_AGENT);
      __hip_atomic_fetch_add(gen, 1u, __ATOMIC_ACQ_REL, __HIP_MEMORY_SCOPE_AGENT);
    } else {
      while (__hip_atomic_load(gen, __ATOMIC_ACQUIRE, __HIP_MEMORY_SCOPE_AGENT) == g) {
        __builtin_amdgcn_s_sleep(8);
      }
    }
  }
  __syncthreads();
}

__global__ __launch_bounds__(256) void cvt_f32_to_bf16(const float* __restrict__ in,
                                                       unsigned short* __restrict__ out, int n) {
  int i = blockIdx.x * 256 + threadIdx.x;
  if (i < n) out[i] = f2bf(in[i]);
}

// ---------------------------------------------------------------------------
// Bidirectional GRU scan. 32 blocks: 0-15 forward, 16-31 backward. Each block
// owns a 64-wide hidden slice; per step it computes a 64x256 pre-activation
// tile (r|z|inn|hn) with WMMA, K folded = x(512)+h(1024), applies the GRU
// nonlinearity, updates h (f32 state + bf16 copy for the next step's GEMM),
// and accumulates the attention dot-product into att_acc[t,b].
// ---------------------------------------------------------------------------
__global__ __launch_bounds__(256) void gru_scan(
    const bf16_t* __restrict__ x_bf,
    const bf16_t* __restrict__ wih_f, const bf16_t* __restrict__ whh_f,
    const bf16_t* __restrict__ wih_b, const bf16_t* __restrict__ whh_b,
    const float* __restrict__ bih_f, const float* __restrict__ bhh_f,
    const float* __restrict__ bih_b, const float* __restrict__ bhh_b,
    const float* __restrict__ att_w,
    float* __restrict__ h32_f, bf16_t* __restrict__ hbf_f,
    float* __restrict__ h32_b, bf16_t* __restrict__ hbf_b,
    float* __restrict__ att_acc, unsigned* __restrict__ bar) {
  const int dir = blockIdx.x >> 4;
  const int c   = blockIdx.x & 15;
  const int j0  = c * 64;

  const bf16_t* wih = dir ? wih_b : wih_f;
  const bf16_t* whh = dir ? whh_b : whh_f;
  const float*  bih = dir ? bih_b : bih_f;
  const float*  bhh = dir ? bhh_b : bhh_f;
  float*  h32 = dir ? h32_b : h32_f;
  bf16_t* hbf = dir ? hbf_b : hbf_f;
  unsigned* cnt = bar + dir * 2;
  unsigned* gen = bar + dir * 2 + 1;

  const int tid  = threadIdx.x;
  const int wave = tid >> 5;
  const int lane = tid & 31;
  const int half = lane >> 4;
  const int l15  = lane & 15;

  __shared__ float smem[64 * 256];  // 64KB: 64 batch x (r|z|inn|hn) slices

  // zero-init this block's slice of h, then sync the direction's 16 blocks
  for (int idx = tid; idx < 64 * 64; idx += 256) {
    int b = idx >> 6, j = idx & 63;
    h32[b * kH + j0 + j] = 0.0f;
    reinterpret_cast<unsigned short*>(hbf)[b * kH + j0 + j] = 0;
  }
  grid_barrier(cnt, gen, 16u);

  // strip_a = wave: waves 0-3 -> r columns, 4-7 -> z columns (K = x + h)
  // strip_b = 8+wave: waves 0-3 -> inn (x only), waves 4-7 -> hn (h only)
  const int ra_row = ((wave < 4) ? (j0 + 16 * wave) : (kH + j0 + 16 * (wave - 4))) + l15;
  const int rb_row = 2 * kH + j0 + 16 * (wave & 3) + l15;

  int brow[4];
#pragma unroll
  for (int mt = 0; mt < 4; ++mt) brow[mt] = 16 * mt + l15;

  for (int t = 0; t < kT; ++t) {
    const int t_x = dir ? (kT - 1 - t) : t;
    v8f accA[4] = {};
    v8f accB[4] = {};

    // ---- x-projection phase (K = kI) ----
    const bf16_t* xrow[4];
#pragma unroll
    for (int mt = 0; mt < 4; ++mt)
      xrow[mt] = x_bf + ((size_t)brow[mt] * kT + t_x) * kI;
    const bf16_t* wra = wih + (size_t)ra_row * kI;
    const bf16_t* wrb = wih + (size_t)rb_row * kI;
    for (int kk = 0; kk < kI; kk += 32) {
      v16bf a[4];
#pragma unroll
      for (int mt = 0; mt < 4; ++mt) a[mt] = load_fragA(xrow[mt], kk, half);
      v16bf bfr = load_fragB(wra, kk, half);
#pragma unroll
      for (int mt = 0; mt < 4; ++mt) accA[mt] = wmma_bf16(a[mt], bfr, accA[mt]);
      if (wave < 4) {  // inn strip (x part of n-gate)
        v16bf bfr2 = load_fragB(wrb, kk, half);
#pragma unroll
        for (int mt = 0; mt < 4; ++mt) accB[mt] = wmma_bf16(a[mt], bfr2, accB[mt]);
      }
    }

    // ---- h-projection phase (K = kH) ----
    const bf16_t* hra = whh + (size_t)ra_row * kH;
    const bf16_t* hrb = whh + (size_t)rb_row * kH;
    for (int kk = 0; kk < kH; kk += 32) {
      v16bf a[4];
#pragma unroll
      for (int mt = 0; mt < 4; ++mt)
        a[mt] = load_fragA(hbf + (size_t)brow[mt] * kH, kk, half);
      v16bf bfr = load_fragB(hra, kk, half);
#pragma unroll
      for (int mt = 0; mt < 4; ++mt) accA[mt] = wmma_bf16(a[mt], bfr, accA[mt]);
      if (wave >= 4) {  // hn strip (h part of n-gate)
        v16bf bfr2 = load_fragB(hrb, kk, half);
#pragma unroll
        for (int mt = 0; mt < 4; ++mt) accB[mt] = wmma_bf16(a[mt], bfr2, accB[mt]);
      }
    }

    // stash accumulators: C/D layout row = 8*half + vgpr, col = lane%16
    const int colA = 16 * wave + l15;
    const int colB = 16 * (8 + wave) + l15;
#pragma unroll
    for (int mt = 0; mt < 4; ++mt) {
#pragma unroll
      for (int r = 0; r < 8; ++r) {
        int row = 16 * mt + 8 * half + r;
        smem[row * 256 + colA] = accA[mt][r];
        smem[row * 256 + colB] = accB[mt][r];
      }
    }
    __syncthreads();

    // GRU gates + state update + attention partial dot-product
    {
      const int b  = tid & 63;
      const int jg = tid >> 6;
      float attv = 0.0f;
#pragma unroll
      for (int jj = 0; jj < 16; ++jj) {
        int j = jg * 16 + jj;
        int J = j0 + j;
        float pr  = smem[b * 256 + j]        + bih[J]          + bhh[J];
        float pz  = smem[b * 256 + 64 + j]   + bih[kH + J]     + bhh[kH + J];
        float pin = smem[b * 256 + 128 + j]  + bih[2 * kH + J];
        float phn = smem[b * 256 + 192 + j]  + bhh[2 * kH + J];
        float r = sigmf(pr);
        float z = sigmf(pz);
        float n = tanhf(pin + r * phn);
        float ho  = h32[b * kH + J];
        float hn2 = (1.0f - z) * n + z * ho;
        h32[b * kH + J] = hn2;
        reinterpret_cast<unsigned short*>(hbf)[b * kH + J] = f2bf(hn2);
        attv += hn2 * att_w[dir * kH + J];
      }
      int t_orig = dir ? (kT - 1 - t) : t;
      atomicAdd(&att_acc[t_orig * kB + b], attv);
    }
    grid_barrier(cnt, gen, 16u);
  }
}

__global__ __launch_bounds__(256) void att_finalize(const float* __restrict__ att_acc,
                                                    const float* __restrict__ fc_b,
                                                    float* __restrict__ att_buf,
                                                    float* __restrict__ out_att) {
  int i = blockIdx.x * 256 + threadIdx.x;
  if (i < kT * kB) {
    int t = i / kB, b = i % kB;
    float v = sigmf(kSigW * (att_acc[i] + fc_b[0]));
    att_buf[i] = v;                // (t, b) for the main scan
    out_att[b * kT + t] = v;       // att.T output (B, T)
  }
}

// ---------------------------------------------------------------------------
// Main TAGM scan: h = a*relu(x@i2h.T + h@h2h.T + b) + (1-a)*h. 16 blocks,
// each owns 64 hidden cols; K folded = x(512)+h(1024); grid barrier per step.
// ---------------------------------------------------------------------------
__global__ __launch_bounds__(256) void main_scan(
    const bf16_t* __restrict__ x_bf, const bf16_t* __restrict__ i2h_w,
    const bf16_t* __restrict__ h2h_w, const float* __restrict__ i2h_b,
    const float* __restrict__ h2h_b, const float* __restrict__ att_buf,
    float* __restrict__ h32, bf16_t* __restrict__ hbf, unsigned* __restrict__ bar) {
  const int j0 = blockIdx.x * 64;
  const int tid = threadIdx.x;
  const int wave = tid >> 5, lane = tid & 31, half = lane >> 4, l15 = lane & 15;
  unsigned* cnt = bar + 4;
  unsigned* gen = bar + 5;

  __shared__ float smem[64 * 64];

  for (int idx = tid; idx < 64 * 64; idx += 256) {
    int b = idx >> 6, j = idx & 63;
    h32[b * kH + j0 + j] = 0.0f;
    reinterpret_cast<unsigned short*>(hbf)[b * kH + j0 + j] = 0;
  }
  grid_barrier(cnt, gen, 16u);

  const int strip = wave >> 1;            // 4 col-strips, 2 waves each
  const int mt0   = (wave & 1) * 2;       // each wave: 2 of 4 M-tiles
  const int nrow  = j0 + 16 * strip + l15;
  const bf16_t* wxr = i2h_w + (size_t)nrow * kI;
  const bf16_t* whr = h2h_w + (size_t)nrow * kH;
  const int b0 = 16 * mt0 + l15;
  const int b1 = 16 * (mt0 + 1) + l15;

  for (int t = 0; t < kT; ++t) {
    v8f acc[2] = {};
    const bf16_t* xr0 = x_bf + ((size_t)b0 * kT + t) * kI;
    const bf16_t* xr1 = x_bf + ((size_t)b1 * kT + t) * kI;
    for (int kk = 0; kk < kI; kk += 32) {
      v16bf bfr = load_fragB(wxr, kk, half);
      acc[0] = wmma_bf16(load_fragA(xr0, kk, half), bfr, acc[0]);
      acc[1] = wmma_bf16(load_fragA(xr1, kk, half), bfr, acc[1]);
    }
    const bf16_t* hr0 = hbf + (size_t)b0 * kH;
    const bf16_t* hr1 = hbf + (size_t)b1 * kH;
    for (int kk = 0; kk < kH; kk += 32) {
      v16bf bfr = load_fragB(whr, kk, half);
      acc[0] = wmma_bf16(load_fragA(hr0, kk, half), bfr, acc[0]);
      acc[1] = wmma_bf16(load_fragA(hr1, kk, half), bfr, acc[1]);
    }
#pragma unroll
    for (int i = 0; i < 2; ++i)
#pragma unroll
      for (int r = 0; r < 8; ++r)
        smem[(16 * (mt0 + i) + 8 * half + r) * 64 + 16 * strip + l15] = acc[i][r];
    __syncthreads();
    {
      int b = tid & 63, jg = tid >> 6;
      float a = att_buf[t * kB + b];
#pragma unroll
      for (int jj = 0; jj < 16; ++jj) {
        int j = jg * 16 + jj, J = j0 + j;
        float tr = smem[b * 64 + j] + i2h_b[J] + h2h_b[J];
        tr = tr > 0.0f ? tr : 0.0f;
        float ho  = h32[b * kH + J];
        float hn2 = a * tr + (1.0f - a) * ho;
        h32[b * kH + J] = hn2;
        reinterpret_cast<unsigned short*>(hbf)[b * kH + J] = f2bf(hn2);
      }
    }
    grid_barrier(cnt, gen, 16u);
  }
}

// Dense layer: out[b,n] = A[b,:] . W[n,:] + bias[n]; M=64, N=64*gridDim.x.
__global__ __launch_bounds__(256) void fc_gemm(
    const bf16_t* __restrict__ A, const bf16_t* __restrict__ W,
    const float* __restrict__ bias, int K, int ldOut,
    float* __restrict__ out_f32, unsigned short* __restrict__ out_bf) {
  const int n0 = blockIdx.x * 64;
  const int tid = threadIdx.x;
  const int wave = tid >> 5, lane = tid & 31, half = lane >> 4, l15 = lane & 15;
  __shared__ float smem[64 * 64];
  const int strip = wave >> 1, mt0 = (wave & 1) * 2;
  const bf16_t* wr  = W + (size_t)(n0 + 16 * strip + l15) * K;
  const bf16_t* ar0 = A + (size_t)(16 * mt0 + l15) * K;
  const bf16_t* ar1 = A + (size_t)(16 * (mt0 + 1) + l15) * K;
  v8f acc[2] = {};
  for (int kk = 0; kk < K; kk += 32) {
    v16bf bfr = load_fragB(wr, kk, half);
    acc[0] = wmma_bf16(load_fragA(ar0, kk, half), bfr, acc[0]);
    acc[1] = wmma_bf16(load_fragA(ar1, kk, half), bfr, acc[1]);
  }
#pragma unroll
  for (int i = 0; i < 2; ++i)
#pragma unroll
    for (int r = 0; r < 8; ++r)
      smem[(16 * (mt0 + i) + 8 * half + r) * 64 + 16 * strip + l15] = acc[i][r];
  __syncthreads();
  int b = tid & 63, jg = tid >> 6;
#pragma unroll
  for (int jj = 0; jj < 16; ++jj) {
    int j = jg * 16 + jj;
    float v = smem[b * 64 + j] + bias[n0 + j];
    if (out_f32) out_f32[(size_t)b * ldOut + n0 + j] = v;
    if (out_bf)  out_bf[(size_t)b * ldOut + n0 + j]  = f2bf(v);
  }
}

extern "C" void kernel_launch(void* const* d_in, const int* in_sizes, int n_in,
                              void* d_out, int out_size, void* d_ws, size_t ws_size,
                              hipStream_t stream) {
  (void)in_sizes; (void)n_in; (void)out_size; (void)ws_size;
  const float* x     = (const float*)d_in[0];
  const float* wih_f = (const float*)d_in[1];
  const float* whh_f = (const float*)d_in[2];
  const float* bih_f = (const float*)d_in[3];
  const float* bhh_f = (const float*)d_in[4];
  const float* wih_b = (const float*)d_in[5];
  const float* whh_b = (const float*)d_in[6];
  const float* bih_b = (const float*)d_in[7];
  const float* bhh_b = (const float*)d_in[8];
  const float* attfw = (const float*)d_in[9];
  const float* attfb = (const float*)d_in[10];
  const float* i2h_w = (const float*)d_in[11];
  const float* i2h_b = (const float*)d_in[12];
  const float* h2h_w = (const float*)d_in[13];
  const float* h2h_b = (const float*)d_in[14];
  const float* fc0_w = (const float*)d_in[15];
  const float* fc0_b = (const float*)d_in[16];
  const float* fc1_w = (const float*)d_in[17];
  const float* fc1_b = (const float*)d_in[18];
  float* out = (float*)d_out;

  char* ws = (char*)d_ws;
  size_t off = 0;
  auto alloc = [&](size_t bytes) {
    size_t o = off; off = (off + bytes + 255) & ~(size_t)255; return o;
  };
  // ~60 MB total workspace
  size_t oAttAcc = alloc((size_t)kT * kB * 4);
  size_t oBar    = alloc(256);
  size_t oXbf    = alloc((size_t)kB * kT * kI * 2);
  size_t oWihF   = alloc((size_t)3 * kH * kI * 2);
  size_t oWhhF   = alloc((size_t)3 * kH * kH * 2);
  size_t oWihB   = alloc((size_t)3 * kH * kI * 2);
  size_t oWhhB   = alloc((size_t)3 * kH * kH * 2);
  size_t oI2h    = alloc((size_t)kH * kI * 2);
  size_t oH2h    = alloc((size_t)kH * kH * 2);
  size_t oFc0    = alloc((size_t)kC * kH * 2);
  size_t oFc1    = alloc((size_t)kH * kH * 2);
  size_t oH32F   = alloc((size_t)kB * kH * 4);
  size_t oHbfF   = alloc((size_t)kB * kH * 2);
  size_t oH32B   = alloc((size_t)kB * kH * 4);
  size_t oHbfB   = alloc((size_t)kB * kH * 2);
  size_t oH32M   = alloc((size_t)kB * kH * 4);
  size_t oHbfM   = alloc((size_t)kB * kH * 2);
  size_t oTmpBf  = alloc((size_t)kB * kC * 2);
  size_t oAttBuf = alloc((size_t)kT * kB * 4);

  // att accumulator + barrier counters (contiguous at ws start) -> zero
  hipMemsetAsync(ws + oAttAcc, 0, (oBar - oAttAcc) + 256, stream);

  auto cvt = [&](const float* src, size_t dstOff, size_t n) {
    cvt_f32_to_bf16<<<(unsigned)((n + 255) / 256), 256, 0, stream>>>(
        src, (unsigned short*)(ws + dstOff), (int)n);
  };
  cvt(x,     oXbf,  (size_t)kB * kT * kI);
  cvt(wih_f, oWihF, (size_t)3 * kH * kI);
  cvt(whh_f, oWhhF, (size_t)3 * kH * kH);
  cvt(wih_b, oWihB, (size_t)3 * kH * kI);
  cvt(whh_b, oWhhB, (size_t)3 * kH * kH);
  cvt(i2h_w, oI2h,  (size_t)kH * kI);
  cvt(h2h_w, oH2h,  (size_t)kH * kH);
  cvt(fc0_w, oFc0,  (size_t)kC * kH);
  cvt(fc1_w, oFc1,  (size_t)kH * kH);

  gru_scan<<<32, 256, 0, stream>>>(
      (const bf16_t*)(ws + oXbf),
      (const bf16_t*)(ws + oWihF), (const bf16_t*)(ws + oWhhF),
      (const bf16_t*)(ws + oWihB), (const bf16_t*)(ws + oWhhB),
      bih_f, bhh_f, bih_b, bhh_b, attfw,
      (float*)(ws + oH32F), (bf16_t*)(ws + oHbfF),
      (float*)(ws + oH32B), (bf16_t*)(ws + oHbfB),
      (float*)(ws + oAttAcc), (unsigned*)(ws + oBar));

  att_finalize<<<(kT * kB + 255) / 256, 256, 0, stream>>>(
      (const float*)(ws + oAttAcc), attfb, (float*)(ws + oAttBuf),
      out + (size_t)kB * kC);

  main_scan<<<16, 256, 0, stream>>>(
      (const bf16_t*)(ws + oXbf), (const bf16_t*)(ws + oI2h),
      (const bf16_t*)(ws + oH2h), i2h_b, h2h_b,
      (const float*)(ws + oAttBuf),
      (float*)(ws + oH32M), (bf16_t*)(ws + oHbfM), (unsigned*)(ws + oBar));

  // out = (h_last @ fc0.T + b0) @ fc1.T + b1
  fc_gemm<<<16, 256, 0, stream>>>(
      (const bf16_t*)(ws + oHbfM), (const bf16_t*)(ws + oFc0), fc0_b, kH, kC,
      nullptr, (unsigned short*)(ws + oTmpBf));
  fc_gemm<<<16, 256, 0, stream>>>(
      (const bf16_t*)(ws + oTmpBf), (const bf16_t*)(ws + oFc1), fc1_b, kC, kH,
      out, nullptr);
}
